// JKNet_22694607192491
// MI455X (gfx1250) — matrix operand order repaired and checked by reading, not compiled, not run on hardware
//
#include <hip/hip_runtime.h>
#include <stdint.h>

// ---------------------------------------------------------------------------
// JKNet (GCN x2 + bidirectional LSTM JumpingKnowledge + APPNP + log_softmax)
// CDNA5 / gfx1250: dense GEMMs via v_wmma_f32_16x16x32_f16 with fully
// unrolled K loops, A slab staged in LDS via async global->LDS copies
// (ASYNCcnt path), ds_load_b128 fragments, B held transposed [N][K] in
// global so fragments are two global_load_b128 each. Scatter propagation via
// global_atomic_add_f32.
// ---------------------------------------------------------------------------

#define Nn   50000
#define Ee   800000
#define INF_ 512
#define HID_ 128
#define OUT_ 64
#define LH_  256     // (4 layers * 128) / 2
#define G4   1024    // 4 * LH_

typedef __attribute__((ext_vector_type(16))) _Float16 v16h;
typedef __attribute__((ext_vector_type(8)))  float    v8f;

union V16 {                      // fragment builder: 2 x 16B -> v16h
    v16h   v;
    float4 f4[2];
};

#if __has_builtin(__builtin_amdgcn_global_load_async_to_lds_b128) && \
    __has_builtin(__builtin_amdgcn_s_wait_asynccnt)
#define HAVE_ASYNC_LDS 1
typedef int v4i_ __attribute__((vector_size(16)));
typedef __attribute__((address_space(1))) v4i_ GlobV4;   // __device__ int4*
typedef __attribute__((address_space(3))) v4i_ LdsV4;    // __shared__ int4*
#endif

// ---------------------------------------------------------------------------
// Tiled WMMA GEMM:  C[M,N] = (accum ? C : 0) + A[M,K] @ BT[N,K]^T + bias
// A f16 row-major, BT f16 = B transposed (BT[n*K+k] == B[k][n]), C f32.
// M%16==0, N%16==0, K in {128,256,512} (template, fully unrolled).
// 256 threads = 8 waves; block tile = 16(M) x 128(N); wave owns one 16x16.
// A slab (16 x K) staged in LDS once, shared by all waves.
// ---------------------------------------------------------------------------
#define TILE_N 128
#define APAD   8                          // f16 pad -> conflict-free ds_load

template <int K>
__global__ void __launch_bounds__(256)
gemm_wmma_f16(const _Float16* __restrict__ A, const _Float16* __restrict__ BT,
              const float* __restrict__ bias, float* __restrict__ C,
              int M, int N, int accum, int relu)
{
    constexpr int KT      = K / 32;       // k-tiles
    constexpr int ASTRIDE = K + APAD;     // padded LDS row stride (f16)
    __shared__ alignas(16) _Float16 As[16 * ASTRIDE];

    const int tid  = threadIdx.x;
    const int lane = tid & 31;
    const int wave = tid >> 5;            // 0..7
    const int nBlockBase = blockIdx.x * TILE_N;
    int NT = N - nBlockBase;
    if (NT > TILE_N) NT = TILE_N;
    const int mBase = blockIdx.y * 16;
    const int nTile = wave * 16;          // wave's N offset inside block tile

    // ---- cooperative stage of A slab [16 x K] into LDS (16B chunks) ----
    constexpr int CHUNKS = 16 * K / 8;    // 8 f16 per chunk
    for (int ch = tid; ch < CHUNKS; ch += 256) {
        const int row  = ch / (K / 8);
        const int coff = (ch % (K / 8)) * 8;
#ifdef HAVE_ASYNC_LDS
        // CDNA5 async copy: cache/memory -> LDS directly, tracked by ASYNCcnt
        __builtin_amdgcn_global_load_async_to_lds_b128(
            (GlobV4*)(uintptr_t)(A + (size_t)(mBase + row) * K + coff),
            (LdsV4*)(As + row * ASTRIDE + coff),
            0, 0);
#else
        *(float4*)(As + row * ASTRIDE + coff) =
            *(const float4*)(A + (size_t)(mBase + row) * K + coff);
#endif
    }
#ifdef HAVE_ASYNC_LDS
    __builtin_amdgcn_s_wait_asynccnt(0);
#endif
    __syncthreads();

    if (nTile >= NT) return;              // wave-uniform; no barriers follow

    const int half = lane >> 4;           // 0 or 1
    const int l16  = lane & 15;
    const int n    = nBlockBase + nTile + l16;

    v8f acc = {};
    if (accum) {
        const int m0 = mBase + half * 8;
        #pragma unroll
        for (int r = 0; r < 8; ++r)
            acc[r] = C[(size_t)(m0 + r) * N + n];
    }

    // Lane base pointers.
    //  A frag (16-bit A 16x32 ISA layout): lane<16 K={kb..kb+7,16+kb..},
    //  kb = half*8 -> two contiguous 16B LDS loads at +0 and +16.
    const _Float16* abase  = As + l16 * ASTRIDE + half * 8;
    //  B frag (16-bit B 32x16): same K order as A: K={kb..kb+7,16+kb..}, kb = half*8.
    const _Float16* btbase = BT + (size_t)n * K + half * 8;
    __builtin_prefetch(btbase, 0, 0);

    #pragma unroll
    for (int kt = 0; kt < KT; ++kt) {
        V16 a, b;
        a.f4[0] = *(const float4*)(abase  + kt * 32);       // ds_load_b128
        a.f4[1] = *(const float4*)(abase  + kt * 32 + 16);  // ds_load_b128
        b.f4[0] = *(const float4*)(btbase + kt * 32);       // global_load_b128
        b.f4[1] = *(const float4*)(btbase + kt * 32 + 16);  // global_load_b128
        acc = __builtin_amdgcn_wmma_f32_16x16x32_f16(
                  false, a.v, false, b.v, (short)0, acc, false, false);
    }

    const int m0 = mBase + half * 8;
    const float bv = bias ? bias[n] : 0.0f;
    #pragma unroll
    for (int r = 0; r < 8; ++r) {
        float v = acc[r] + bv;
        if (relu) v = fmaxf(v, 0.0f);
        C[(size_t)(m0 + r) * N + n] = v;
    }
}

// ---------------------------------------------------------------------------
// Small helper kernels
// ---------------------------------------------------------------------------
__global__ void cvt_f32_to_f16(const float* __restrict__ src,
                               _Float16* __restrict__ dst, int n)
{
    int i = blockIdx.x * 256 + threadIdx.x;
    if (i < n) dst[i] = (_Float16)src[i];
}

// dst[c*R + r] = src[r*C + c]  (f32 [R,C] -> f16 [C,R]; builds BT from [K,N])
__global__ void transpose_cvt_f16(const float* __restrict__ src,
                                  _Float16* __restrict__ dst, int R, int C)
{
    int i = blockIdx.x * 256 + threadIdx.x;
    if (i < R * C) {
        int r = i / C, c = i - r * C;
        dst[(size_t)c * R + r] = (_Float16)src[i];
    }
}

__global__ void fill_f32(float* p, float v, int n)
{
    int i = blockIdx.x * 256 + threadIdx.x;
    if (i < n) p[i] = v;
}

__global__ void deg_accum(const int* __restrict__ col,
                          const float* __restrict__ w,
                          float* __restrict__ deg, int E)
{
    int e = blockIdx.x * 256 + threadIdx.x;
    if (e < E) atomicAdd(&deg[col[e]], w[e]);
}

__global__ void deg_finish(const float* __restrict__ deg,
                           float* __restrict__ dis, int n)
{
    int i = blockIdx.x * 256 + threadIdx.x;
    if (i < n) {
        float d = deg[i] + 1.0f;            // +1 = self loop weight
        dis[i] = (d > 0.0f) ? rsqrtf(fmaxf(d, 1e-30f)) : 0.0f;
    }
}

// out[col] += dis[row]*w*dis[col] * x[row]  for each edge (H==128 features)
__global__ void propagate_scatter(const float* __restrict__ x,
                                  const int* __restrict__ rows,
                                  const int* __restrict__ cols,
                                  const float* __restrict__ w,
                                  const float* __restrict__ dis,
                                  float* __restrict__ out, int E)
{
    int e = blockIdx.x;
    if (e >= E) return;
    int t = threadIdx.x;                    // 0..127
    int r = rows[e], c = cols[e];
    float coef = dis[r] * w[e] * dis[c];
    atomicAdd(&out[(size_t)c * HID_ + t], coef * x[(size_t)r * HID_ + t]);
}

// out = [relu](prop + dis[node]^2 * xw + bias)   (self-loop term fused)
__global__ void finish_prop(const float* __restrict__ prop,
                            const float* __restrict__ xw,
                            const float* __restrict__ dis,
                            const float* __restrict__ bias,
                            float* __restrict__ out, int total, int relu)
{
    int idx = blockIdx.x * 256 + threadIdx.x;
    if (idx >= total) return;
    int node = idx >> 7, f = idx & 127;
    float d = dis[node];
    float v = prop[idx] + d * d * xw[idx] + (bias ? bias[f] : 0.0f);
    if (relu) v = fmaxf(v, 0.0f);
    out[idx] = v;
}

__device__ __forceinline__ float sigmoidf_(float x)
{
    return 1.0f / (1.0f + __expf(-x));
}

// gates g: [N,1024] = x@Wih.T (+ h@Whh.T); gate order i,f,g,o (chunks of 256)
__global__ void lstm_pointwise(const float* __restrict__ g,
                               const float* __restrict__ bih,
                               const float* __restrict__ bhh,
                               float* __restrict__ h, float* __restrict__ c,
                               float* __restrict__ hstore, int total)
{
    int idx = blockIdx.x * 256 + threadIdx.x;
    if (idx >= total) return;
    int node = idx >> 8, j = idx & 255;
    const float* gn = g + (size_t)node * G4;
    float gi = gn[j]       + bih[j]       + bhh[j];
    float gf = gn[256 + j] + bih[256 + j] + bhh[256 + j];
    float gg = gn[512 + j] + bih[512 + j] + bhh[512 + j];
    float go = gn[768 + j] + bih[768 + j] + bhh[768 + j];
    float cc = sigmoidf_(gf) * c[idx] + sigmoidf_(gi) * tanhf(gg);
    float hh = sigmoidf_(go) * tanhf(cc);
    c[idx] = cc;
    h[idx] = hh;
    hstore[idx] = hh;
}

// Per-node: alpha_l = attb + <[hf_l,hb_l], attW>; softmax over 2 layers;
// xjk = a0*x1 + a1*x2.  One 128-thread block per node.
__global__ void jk_attention(const float* __restrict__ hf,
                             const float* __restrict__ hb,
                             const float* __restrict__ attW,
                             const float* __restrict__ attb,
                             const float* __restrict__ x1,
                             const float* __restrict__ x2,
                             float* __restrict__ xjk)
{
    __shared__ float red[128];
    int node = blockIdx.x, t = threadIdx.x;
    float alpha[2];
    #pragma unroll
    for (int l = 0; l < 2; ++l) {
        const float* f = hf + ((size_t)l * Nn + node) * LH_;
        const float* b = hb + ((size_t)l * Nn + node) * LH_;
        float s = 0.0f;
        for (int j = t; j < LH_; j += 128)
            s += f[j] * attW[j] + b[j] * attW[LH_ + j];
        red[t] = s; __syncthreads();
        for (int off = 64; off > 0; off >>= 1) {
            if (t < off) red[t] += red[t + off];
            __syncthreads();
        }
        alpha[l] = red[0] + attb[0];
        __syncthreads();
    }
    float m  = fmaxf(alpha[0], alpha[1]);
    float e0 = __expf(alpha[0] - m), e1 = __expf(alpha[1] - m);
    float inv = 1.0f / (e0 + e1);
    size_t p = (size_t)node * HID_ + t;
    xjk[p] = (e0 * inv) * x1[p] + (e1 * inv) * x2[p];
}

// Row-wise log_softmax over 64 logits; one 64-thread block per node.
__global__ void log_softmax64(const float* __restrict__ logits,
                              float* __restrict__ out)
{
    __shared__ float red[64];
    int node = blockIdx.x, t = threadIdx.x;
    float v = logits[(size_t)node * OUT_ + t];
    red[t] = v; __syncthreads();
    for (int off = 32; off > 0; off >>= 1) {
        if (t < off) red[t] = fmaxf(red[t], red[t + off]);
        __syncthreads();
    }
    float m = red[0]; __syncthreads();
    red[t] = __expf(v - m); __syncthreads();
    for (int off = 32; off > 0; off >>= 1) {
        if (t < off) red[t] += red[t + off];
        __syncthreads();
    }
    out[(size_t)node * OUT_ + t] = v - m - logf(red[0]);
}

// ---------------------------------------------------------------------------
// Host orchestration
// ---------------------------------------------------------------------------
static inline dim3 gemm_grid(int M, int N)
{
    return dim3((N + TILE_N - 1) / TILE_N, M / 16);
}

extern "C" void kernel_launch(void* const* d_in, const int* in_sizes, int n_in,
                              void* d_out, int out_size, void* d_ws, size_t ws_size,
                              hipStream_t stream)
{
    (void)in_sizes; (void)n_in; (void)out_size; (void)ws_size;

    const float* x      = (const float*)d_in[0];
    const int*   eidx   = (const int*)  d_in[1];
    const float* ew     = (const float*)d_in[2];
    const float* W1     = (const float*)d_in[3];
    const float* b1     = (const float*)d_in[4];
    const float* W2     = (const float*)d_in[5];
    const float* b2     = (const float*)d_in[6];
    const float* Wih_f  = (const float*)d_in[7];
    const float* Whh_f  = (const float*)d_in[8];
    const float* bih_f  = (const float*)d_in[9];
    const float* bhh_f  = (const float*)d_in[10];
    const float* Wih_b  = (const float*)d_in[11];
    const float* Whh_b  = (const float*)d_in[12];
    const float* bih_b  = (const float*)d_in[13];
    const float* bhh_b  = (const float*)d_in[14];
    const float* attW   = (const float*)d_in[15];
    const float* attb   = (const float*)d_in[16];
    const float* linW   = (const float*)d_in[17];
    const float* linb   = (const float*)d_in[18];

    const int* rows = eidx;          // edge_index[0]
    const int* cols = eidx + Ee;     // edge_index[1]

    // ---- workspace bump allocator (256B aligned) ----
    char* ws = (char*)d_ws;
    size_t off = 0;
    auto alloc = [&](size_t bytes) -> void* {
        void* p = ws + off;
        off = (off + bytes + 255) & ~(size_t)255;
        return p;
    };

    float*    deg    = (float*)   alloc((size_t)Nn * 4);
    float*    dis    = (float*)   alloc((size_t)Nn * 4);
    _Float16* x16    = (_Float16*)alloc((size_t)Nn * INF_ * 2);
    _Float16* W1T    = (_Float16*)alloc((size_t)INF_ * HID_ * 2);   // [128][512]
    _Float16* W2T    = (_Float16*)alloc((size_t)HID_ * HID_ * 2);   // [128][128]
    _Float16* Wih16f = (_Float16*)alloc((size_t)G4 * HID_ * 2);     // [1024][128]
    _Float16* Whh16f = (_Float16*)alloc((size_t)G4 * LH_ * 2);      // [1024][256]
    _Float16* Wih16b = (_Float16*)alloc((size_t)G4 * HID_ * 2);
    _Float16* Whh16b = (_Float16*)alloc((size_t)G4 * LH_ * 2);
    _Float16* linWT  = (_Float16*)alloc((size_t)HID_ * OUT_ * 2);   // [64][128]
    float*    bufA   = (float*)   alloc((size_t)Nn * HID_ * 4);     // gemm out / xjk
    float*    bufB   = (float*)   alloc((size_t)Nn * HID_ * 4);     // propagate out
    float*    x1     = (float*)   alloc((size_t)Nn * HID_ * 4);
    float*    x2     = (float*)   alloc((size_t)Nn * HID_ * 4);
    _Float16* x1h    = (_Float16*)alloc((size_t)Nn * HID_ * 2);
    _Float16* x2h    = (_Float16*)alloc((size_t)Nn * HID_ * 2);
    float*    gx     = (float*)   alloc((size_t)Nn * G4 * 4);       // LSTM gates
    float*    hbuf   = (float*)   alloc((size_t)Nn * LH_ * 4);
    float*    cbuf   = (float*)   alloc((size_t)Nn * LH_ * 4);
    _Float16* h16    = (_Float16*)alloc((size_t)Nn * LH_ * 2);
    float*    hf     = (float*)   alloc((size_t)2 * Nn * LH_ * 4);
    float*    hb     = (float*)   alloc((size_t)2 * Nn * LH_ * 4);

    auto blk1 = [](int n) { return dim3((n + 255) / 256); };

    // ---- gcn_norm: degrees (incl. self loop) -> dis = rsqrt(deg) ----
    fill_f32<<<blk1(Nn), 256, 0, stream>>>(deg, 0.0f, Nn);
    deg_accum<<<blk1(Ee), 256, 0, stream>>>(cols, ew, deg, Ee);
    deg_finish<<<blk1(Nn), 256, 0, stream>>>(deg, dis, Nn);

    // ---- operand prep: A matrices -> f16; B matrices -> f16 transposed [N][K]
    //      (Wih/Whh are stored [gates, K] == [N][K] already: plain convert) ----
    cvt_f32_to_f16<<<blk1(Nn * INF_), 256, 0, stream>>>(x, x16, Nn * INF_);
    transpose_cvt_f16<<<blk1(INF_ * HID_), 256, 0, stream>>>(W1, W1T, INF_, HID_);
    transpose_cvt_f16<<<blk1(HID_ * HID_), 256, 0, stream>>>(W2, W2T, HID_, HID_);
    transpose_cvt_f16<<<blk1(HID_ * OUT_), 256, 0, stream>>>(linW, linWT, HID_, OUT_);
    cvt_f32_to_f16<<<blk1(G4 * HID_), 256, 0, stream>>>(Wih_f, Wih16f, G4 * HID_);
    cvt_f32_to_f16<<<blk1(G4 * LH_),  256, 0, stream>>>(Whh_f, Whh16f, G4 * LH_);
    cvt_f32_to_f16<<<blk1(G4 * HID_), 256, 0, stream>>>(Wih_b, Wih16b, G4 * HID_);
    cvt_f32_to_f16<<<blk1(G4 * LH_),  256, 0, stream>>>(Whh_b, Whh16b, G4 * LH_);

    // ---- GCN layer 1: x@W1 -> propagate -> +b1, ReLU ----
    gemm_wmma_f16<INF_><<<gemm_grid(Nn, HID_), 256, 0, stream>>>(
        x16, W1T, nullptr, bufA, Nn, HID_, 0, 0);
    fill_f32<<<blk1(Nn * HID_), 256, 0, stream>>>(bufB, 0.0f, Nn * HID_);
    propagate_scatter<<<dim3(Ee), 128, 0, stream>>>(bufA, rows, cols, ew, dis, bufB, Ee);
    finish_prop<<<blk1(Nn * HID_), 256, 0, stream>>>(bufB, bufA, dis, b1, x1, Nn * HID_, 1);

    // ---- GCN layer 2 ----
    cvt_f32_to_f16<<<blk1(Nn * HID_), 256, 0, stream>>>(x1, x1h, Nn * HID_);
    gemm_wmma_f16<HID_><<<gemm_grid(Nn, HID_), 256, 0, stream>>>(
        x1h, W2T, nullptr, bufA, Nn, HID_, 0, 0);
    fill_f32<<<blk1(Nn * HID_), 256, 0, stream>>>(bufB, 0.0f, Nn * HID_);
    propagate_scatter<<<dim3(Ee), 128, 0, stream>>>(bufA, rows, cols, ew, dis, bufB, Ee);
    finish_prop<<<blk1(Nn * HID_), 256, 0, stream>>>(bufB, bufA, dis, b2, x2, Nn * HID_, 1);
    cvt_f32_to_f16<<<blk1(Nn * HID_), 256, 0, stream>>>(x2, x2h, Nn * HID_);

    const int nh = Nn * LH_;

    // ---- forward LSTM over layers [x1, x2] ----
    fill_f32<<<blk1(nh), 256, 0, stream>>>(cbuf, 0.0f, nh);
    gemm_wmma_f16<HID_><<<gemm_grid(Nn, G4), 256, 0, stream>>>(
        x1h, Wih16f, nullptr, gx, Nn, G4, 0, 0);                 // h0 == 0
    lstm_pointwise<<<blk1(nh), 256, 0, stream>>>(gx, bih_f, bhh_f,
                                                 hbuf, cbuf, hf, nh);
    cvt_f32_to_f16<<<blk1(nh), 256, 0, stream>>>(hbuf, h16, nh);
    gemm_wmma_f16<HID_><<<gemm_grid(Nn, G4), 256, 0, stream>>>(
        x2h, Wih16f, nullptr, gx, Nn, G4, 0, 0);
    gemm_wmma_f16<LH_><<<gemm_grid(Nn, G4), 256, 0, stream>>>(
        h16, Whh16f, nullptr, gx, Nn, G4, 1, 0);                 // accumulate
    lstm_pointwise<<<blk1(nh), 256, 0, stream>>>(gx, bih_f, bhh_f,
                                                 hbuf, cbuf, hf + (size_t)nh, nh);

    // ---- backward LSTM over layers [x2, x1] (stored in forward layer order) --
    fill_f32<<<blk1(nh), 256, 0, stream>>>(cbuf, 0.0f, nh);
    gemm_wmma_f16<HID_><<<gemm_grid(Nn, G4), 256, 0, stream>>>(
        x2h, Wih16b, nullptr, gx, Nn, G4, 0, 0);
    lstm_pointwise<<<blk1(nh), 256, 0, stream>>>(gx, bih_b, bhh_b,
                                                 hbuf, cbuf, hb + (size_t)nh, nh);
    cvt_f32_to_f16<<<blk1(nh), 256, 0, stream>>>(hbuf, h16, nh);
    gemm_wmma_f16<HID_><<<gemm_grid(Nn, G4), 256, 0, stream>>>(
        x1h, Wih16b, nullptr, gx, Nn, G4, 0, 0);
    gemm_wmma_f16<LH_><<<gemm_grid(Nn, G4), 256, 0, stream>>>(
        h16, Whh16b, nullptr, gx, Nn, G4, 1, 0);
    lstm_pointwise<<<blk1(nh), 256, 0, stream>>>(gx, bih_b, bhh_b,
                                                 hbuf, cbuf, hb, nh);

    // ---- JK attention -> xjk (bufA) ----
    jk_attention<<<dim3(Nn), 128, 0, stream>>>(hf, hb, attW, attb, x1, x2, bufA);

    // ---- APPNP (K=1, alpha=0): one propagate ----
    fill_f32<<<blk1(Nn * HID_), 256, 0, stream>>>(bufB, 0.0f, Nn * HID_);
    propagate_scatter<<<dim3(Ee), 128, 0, stream>>>(bufA, rows, cols, ew, dis, bufB, Ee);
    finish_prop<<<blk1(Nn * HID_), 256, 0, stream>>>(bufB, bufA, dis, nullptr,
                                                     x1 /*reuse as xp*/, Nn * HID_, 0);

    // ---- final linear + log_softmax ----
    cvt_f32_to_f16<<<blk1(Nn * HID_), 256, 0, stream>>>(x1, x1h, Nn * HID_);
    gemm_wmma_f16<HID_><<<gemm_grid(Nn, OUT_), 256, 0, stream>>>(
        x1h, linWT, linb, bufB /*logits*/, Nn, OUT_, 0, 0);
    log_softmax64<<<dim3(Nn), 64, 0, stream>>>(bufB, (float*)d_out);
}
